// GlobalLocalModel_64768106824145
// MI455X (gfx1250) — matrix operand order repaired and checked by reading, not compiled, not run on hardware
//
#include <hip/hip_runtime.h>

typedef __attribute__((ext_vector_type(16))) _Float16 v16h;
typedef __attribute__((ext_vector_type(8)))  _Float16 v8h;
typedef __attribute__((ext_vector_type(8)))  float    v8f;

#define FH 128

__device__ __forceinline__ float eluf(float x) { return x > 0.f ? x : (__expf(x) - 1.f); }

// ---------------------------------------------------------------------------
// Pack a 256x128 f32 weight into f16 WMMA B-fragment tile order.
// Tile (ktile 0..7, ntile 0..7); per tile 32 lanes x 16 halves.
// b[i] of lane L = W[ktile*32 + 16*(L>=16) + i][ntile*16 + (L&15)]
// ---------------------------------------------------------------------------
__global__ void k_pack_w(const float* __restrict__ W, _Float16* __restrict__ P) {
  int d = blockIdx.x * blockDim.x + threadIdx.x;
  if (d >= 64 * 512) return;
  int tile = d >> 9, r = d & 511;
  int lane = r >> 4, i = r & 15;
  int ktile = tile >> 3, ntile = tile & 7;
  int k = ktile * 32 + ((lane >> 4) << 4) + i;
  int n = ntile * 16 + (lane & 15);
  P[d] = (_Float16)W[k * FH + n];
}

__global__ void k_elu16(const float* __restrict__ X, _Float16* __restrict__ E, int cnt) {
  int i = blockIdx.x * blockDim.x + threadIdx.x;
  if (i < cnt) E[i] = (_Float16)eluf(X[i]);
}

// COO SpMM, rows sorted ascending. One wave per output row; lane covers 4 feats.
__global__ void k_spmm(const int* __restrict__ rows, const int* __restrict__ cols,
                       const float* __restrict__ vals, int ne,
                       const _Float16* __restrict__ E, _Float16* __restrict__ S, int n) {
  int wave = blockIdx.x * (blockDim.x >> 5) + (threadIdx.x >> 5);
  int lane = threadIdx.x & 31;
  if (wave >= n) return;
  int lo = 0, hi = ne;
  while (lo < hi) { int m = (lo + hi) >> 1; if (rows[m] < wave) lo = m + 1; else hi = m; }
  int s0 = lo; hi = ne;
  while (lo < hi) { int m = (lo + hi) >> 1; if (rows[m] <= wave) lo = m + 1; else hi = m; }
  int s1 = lo;
  float a0 = 0, a1 = 0, a2 = 0, a3 = 0;
  for (int e = s0; e < s1; ++e) {
    float v = vals[e];
    const _Float16* ep = E + (size_t)cols[e] * FH + lane;
    a0 += v * (float)ep[0];  a1 += v * (float)ep[32];
    a2 += v * (float)ep[64]; a3 += v * (float)ep[96];
  }
  _Float16* sp = S + (size_t)wave * FH + lane;
  sp[0] = (_Float16)a0; sp[32] = (_Float16)a1; sp[64] = (_Float16)a2; sp[96] = (_Float16)a3;
}

__device__ __forceinline__ v16h load_a_frag(const _Float16* ap) {
  v8h a0 = *(const v8h*)ap;
  v8h a1 = *(const v8h*)(ap + 8);
  return __builtin_shufflevector(a0, a1, 0,1,2,3,4,5,6,7,8,9,10,11,12,13,14,15);
}

// ---------------------------------------------------------------------------
// out[M,128] = concat(E, S|broadcast(AVG)) (M,256) @ Wpacked (256,128) + bias (+res)
// Block = 256 threads (8 waves); wave w -> N-tile w. Each block computes 32 rows
// (two 16-row M-tiles): B fragments loaded once per K-step and reused for two
// independent WMMA accumulator chains (better XDL pipelining, half the W traffic).
// A layout (16-bit 16x32): lanes<16: K=kk+0..15; lanes>=16: K=kk+16..31.
// ---------------------------------------------------------------------------
__global__ void k_gemm(const _Float16* __restrict__ E, const _Float16* __restrict__ S,
                       const _Float16* __restrict__ AVG,
                       const _Float16* __restrict__ WP, const float* __restrict__ bias,
                       const float* __restrict__ res, float* __restrict__ out) {
  int lane = threadIdx.x & 31;
  int ntile = threadIdx.x >> 5;
  int mbase = blockIdx.x * 32;
  int hi = lane >> 4;
  int mrow0 = mbase + (lane & 15);
  int mrow1 = mrow0 + 16;
  v8f c0 = {};
  v8f c1 = {};
#pragma unroll
  for (int kt = 0; kt < 8; ++kt) {
    int kk = kt * 32;
    const _Float16* bp = WP + (size_t)(kt * 8 + ntile) * 512 + lane * 16;
    v8h b0 = *(const v8h*)bp;
    v8h b1 = *(const v8h*)(bp + 8);
    v16h b = __builtin_shufflevector(b0, b1, 0,1,2,3,4,5,6,7,8,9,10,11,12,13,14,15);
    v16h a0, a1;
    if (kk < 128) {
      a0 = load_a_frag(E + (size_t)mrow0 * FH + kk + hi * 16);
      a1 = load_a_frag(E + (size_t)mrow1 * FH + kk + hi * 16);
    } else if (S) {
      a0 = load_a_frag(S + (size_t)mrow0 * FH + (kk - 128) + hi * 16);
      a1 = load_a_frag(S + (size_t)mrow1 * FH + (kk - 128) + hi * 16);
    } else {
      a0 = load_a_frag(AVG + (kk - 128) + hi * 16);
      a1 = a0;
    }
    c0 = __builtin_amdgcn_wmma_f32_16x16x32_f16(false, a0, false, b, (short)0, c0, false, false);
    c1 = __builtin_amdgcn_wmma_f32_16x16x32_f16(false, a1, false, b, (short)0, c1, false, false);
  }
  int ncol = ntile * 16 + (lane & 15);
  int mtop = mbase + hi * 8;
  float bv = bias[ncol];
#pragma unroll
  for (int i = 0; i < 8; ++i) {
    size_t o0 = (size_t)(mtop + i) * FH + ncol;
    size_t o1 = (size_t)(mtop + 16 + i) * FH + ncol;
    float v0 = c0[i] + bv;
    float v1 = c1[i] + bv;
    if (res) { v0 += res[o0]; v1 += res[o1]; }
    out[o0] = v0;
    out[o1] = v1;
  }
}

__global__ void k_mask_sum(const float* __restrict__ mask, float* __restrict__ den, int n) {
  __shared__ float sm[256];
  float s = 0;
  for (int i = threadIdx.x; i < n; i += blockDim.x) s += mask[i];
  sm[threadIdx.x] = s; __syncthreads();
  for (int st = 128; st > 0; st >>= 1) {
    if (threadIdx.x < st) sm[threadIdx.x] += sm[threadIdx.x + st];
    __syncthreads();
  }
  if (threadIdx.x == 0) den[0] = sm[0];
}

__global__ void k_avg(const _Float16* __restrict__ E, const float* __restrict__ mask,
                      const float* __restrict__ den, _Float16* __restrict__ AVG, int n) {
  int f = blockIdx.x;  // 0..127
  __shared__ float sm[256];
  float s = 0;
  for (int i = threadIdx.x; i < n; i += blockDim.x)
    s += mask[i] * (float)E[(size_t)i * FH + f];
  sm[threadIdx.x] = s; __syncthreads();
  for (int st = 128; st > 0; st >>= 1) {
    if (threadIdx.x < st) sm[threadIdx.x] += sm[threadIdx.x + st];
    __syncthreads();
  }
  if (threadIdx.x == 0) AVG[f] = (_Float16)(sm[0] / den[0]);
}

__global__ void k_conv_in(const float* __restrict__ in3, const float* __restrict__ W,
                          const float* __restrict__ b, float* __restrict__ out, int n) {
  int idx = blockIdx.x * blockDim.x + threadIdx.x;
  if (idx >= n * FH) return;
  int node = idx >> 7, f = idx & 127;
  out[idx] = in3[node * 3] * W[f] + in3[node * 3 + 1] * W[FH + f] +
             in3[node * 3 + 2] * W[2 * FH + f] + b[f];
}

__global__ void k_maxpool(const float* __restrict__ X, float* __restrict__ O, int nout) {
  int idx = blockIdx.x * blockDim.x + threadIdx.x;
  if (idx >= nout * FH) return;
  int node = idx >> 7, f = idx & 127;
  O[idx] = fmaxf(X[(size_t)(2 * node) * FH + f], X[(size_t)(2 * node + 1) * FH + f]);
}

__global__ void k_upadd(const float* __restrict__ X, const float* __restrict__ D,
                        float* __restrict__ O, int nout) {
  int idx = blockIdx.x * blockDim.x + threadIdx.x;
  if (idx >= nout * FH) return;
  int node = idx >> 7, f = idx & 127;
  O[idx] = X[(size_t)(node >> 1) * FH + f] + D[idx];
}

__global__ void k_head_g(const float* __restrict__ X, const float* __restrict__ W,
                         const float* __restrict__ b, const float* __restrict__ in3,
                         float* __restrict__ swg, int n) {
  int idx = blockIdx.x * blockDim.x + threadIdx.x;
  if (idx >= n * 2) return;
  int node = idx >> 1, ch = idx & 1;
  float s = 0;
  for (int k = 0; k < FH; ++k) s += eluf(X[(size_t)node * FH + k]) * W[k * 2 + ch];
  swg[idx] = s + b[ch] + in3[node * 3 + ch];
}

__global__ void k_head_l(const float* __restrict__ Y, const float* __restrict__ W,
                         const float* __restrict__ b, const float* __restrict__ in3,
                         float* __restrict__ sl, int n) {
  int node = blockIdx.x * blockDim.x + threadIdx.x;
  if (node >= n) return;
  float s = 0;
  for (int k = 0; k < FH; ++k) s += eluf(Y[(size_t)node * FH + k]) * W[k];
  sl[node] = s + b[0] + in3[node * 3];
}

__global__ void k_final(const float* __restrict__ swg, const float* __restrict__ sl,
                        float* __restrict__ out, int n) {
  int node = blockIdx.x * blockDim.x + threadIdx.x;
  if (node >= n) return;
  float sg = swg[node * 2];
  float slv = sl[node];
  float wg = 1.f / (1.f + __expf(-sg));
  out[node] = sg;
  out[n + node] = slv;
  out[2 * n + node] = wg * sg + (1.f - wg) * slv;
}

// ---------------------------------------------------------------------------
extern "C" void kernel_launch(void* const* d_in, const int* in_sizes, int n_in,
                              void* d_out, int out_size, void* d_ws, size_t ws_size,
                              hipStream_t stream) {
  const float* inputs = (const float*)d_in[0];
  const float* mask1  = (const float*)d_in[2];
  const int*   lap_rows[4] = {(const int*)d_in[3], (const int*)d_in[6], (const int*)d_in[9], (const int*)d_in[12]};
  const int*   lap_cols[4] = {(const int*)d_in[4], (const int*)d_in[7], (const int*)d_in[10], (const int*)d_in[13]};
  const float* lap_vals[4] = {(const float*)d_in[5], (const float*)d_in[8], (const float*)d_in[11], (const float*)d_in[14]};
  int lap_ne[4] = {in_sizes[3], in_sizes[6], in_sizes[9], in_sizes[12]};
  const int*   L_rows = (const int*)d_in[15];
  const int*   L_cols = (const int*)d_in[16];
  const float* L_vals = (const float*)d_in[17];
  int L_ne = in_sizes[15];
  const float* g_conv1_W = (const float*)d_in[18];
  const float* g_conv1_b = (const float*)d_in[19];
  const float* g_down_W  = (const float*)d_in[20];
  const float* g_down_b  = (const float*)d_in[21];
  const float* g_lap0_W  = (const float*)d_in[22];
  const float* g_lap0_b  = (const float*)d_in[23];
  const float* g_up_W    = (const float*)d_in[24];
  const float* g_up_b    = (const float*)d_in[25];
  const float* g_conv2_W = (const float*)d_in[26];
  const float* g_conv2_b = (const float*)d_in[27];
  const float* l_conv1_W = (const float*)d_in[28];
  const float* l_conv1_b = (const float*)d_in[29];
  const float* l_rn_W    = (const float*)d_in[30];
  const float* l_rn_b    = (const float*)d_in[31];
  const float* l_conv2_W = (const float*)d_in[32];
  const float* l_conv2_b = (const float*)d_in[33];

  const int N = in_sizes[0] / 3;  // 16384
  char* ws = (char*)d_ws;
  const size_t MB = 1024 * 1024;
  _Float16* WPACK = (_Float16*)(ws);                  // 44*32768*2 = 2.88 MB
  float*    XA    = (float*)(ws + 3 * MB);            // 8 MB
  float*    XB    = (float*)(ws + 11 * MB);           // 8 MB
  float*    INP   = (float*)(ws + 19 * MB);           // 8 MB
  _Float16* E16   = (_Float16*)(ws + 27 * MB);        // 4 MB
  _Float16* S16   = (_Float16*)(ws + 31 * MB);        // 4 MB
  float*    D0    = (float*)(ws + 35 * MB);           // 8 MB
  float*    D1    = (float*)(ws + 43 * MB);           // 4 MB
  float*    D2    = (float*)(ws + 47 * MB);           // 2 MB
  float*    SWG   = (float*)(ws + 49 * MB);           // 128 KB
  float*    SL    = (float*)(ws + 49 * MB + 256 * 1024);
  _Float16* AVG   = (_Float16*)(ws + 49 * MB + 512 * 1024);
  float*    DEN   = (float*)(ws + 49 * MB + 512 * 1024 + 1024);

  // --- pack all 44 GEMM weight matrices (slots: gdown 0-5, glap0 6-7, gup 8-13, lrn 14-43)
  int slot = 0;
  for (int m = 0; m < 6; ++m)  k_pack_w<<<128, 256, 0, stream>>>(g_down_W + (size_t)m * 256 * FH, WPACK + (size_t)(slot++) * 32768);
  for (int m = 0; m < 2; ++m)  k_pack_w<<<128, 256, 0, stream>>>(g_lap0_W + (size_t)m * 256 * FH, WPACK + (size_t)(slot++) * 32768);
  for (int m = 0; m < 6; ++m)  k_pack_w<<<128, 256, 0, stream>>>(g_up_W   + (size_t)m * 256 * FH, WPACK + (size_t)(slot++) * 32768);
  for (int m = 0; m < 30; ++m) k_pack_w<<<128, 256, 0, stream>>>(l_rn_W   + (size_t)m * 256 * FH, WPACK + (size_t)(slot++) * 32768);

  auto lap_layer = [&](float* x, int n, const int* rws, const int* cls, const float* vls,
                       int ne, int w, const float* bp, const float* res) {
    int cnt = n * FH;
    k_elu16<<<(cnt + 255) / 256, 256, 0, stream>>>(x, E16, cnt);
    k_spmm<<<(n + 7) / 8, 256, 0, stream>>>(rws, cls, vls, ne, E16, S16, n);
    k_gemm<<<n / 32, 256, 0, stream>>>(E16, S16, (const _Float16*)nullptr,
                                       WPACK + (size_t)w * 32768, bp, res, x);
  };
  auto lap_block = [&](float* x, int n, int li, int w, const float* b0) {
    hipMemcpyAsync(INP, x, (size_t)n * FH * 4, hipMemcpyDeviceToDevice, stream);
    lap_layer(x, n, lap_rows[li], lap_cols[li], lap_vals[li], lap_ne[li], w, b0, nullptr);
    lap_layer(x, n, lap_rows[li], lap_cols[li], lap_vals[li], lap_ne[li], w + 1, b0 + FH, INP);
  };
  auto avg_layer = [&](float* x, int n, int w, const float* bp, const float* res) {
    int cnt = n * FH;
    k_elu16<<<(cnt + 255) / 256, 256, 0, stream>>>(x, E16, cnt);
    k_avg<<<128, 256, 0, stream>>>(E16, mask1, DEN, AVG, n);
    k_gemm<<<n / 32, 256, 0, stream>>>(E16, (const _Float16*)nullptr, AVG,
                                       WPACK + (size_t)w * 32768, bp, res, x);
  };

  // ================= global branch (U-Net) =================
  k_conv_in<<<(N * FH + 255) / 256, 256, 0, stream>>>(inputs, g_conv1_W, g_conv1_b, XA, N);
  float* cur = XA; float* alt = XB;
  // down k=0, lap3, n=N
  hipMemcpyAsync(D0, cur, (size_t)N * FH * 4, hipMemcpyDeviceToDevice, stream);
  lap_block(cur, N, 3, 0, g_down_b + 0 * FH);
  k_maxpool<<<((N / 2) * FH + 255) / 256, 256, 0, stream>>>(cur, alt, N / 2);
  { float* t = cur; cur = alt; alt = t; }
  // down k=1, lap2, n=N/2
  hipMemcpyAsync(D1, cur, (size_t)(N / 2) * FH * 4, hipMemcpyDeviceToDevice, stream);
  lap_block(cur, N / 2, 2, 2, g_down_b + 2 * FH);
  k_maxpool<<<((N / 4) * FH + 255) / 256, 256, 0, stream>>>(cur, alt, N / 4);
  { float* t = cur; cur = alt; alt = t; }
  // down k=2, lap1, n=N/4
  hipMemcpyAsync(D2, cur, (size_t)(N / 4) * FH * 4, hipMemcpyDeviceToDevice, stream);
  lap_block(cur, N / 4, 1, 4, g_down_b + 4 * FH);
  k_maxpool<<<((N / 8) * FH + 255) / 256, 256, 0, stream>>>(cur, alt, N / 8);
  { float* t = cur; cur = alt; alt = t; }
  // bottom, lap0, n=N/8
  lap_block(cur, N / 8, 0, 6, g_lap0_b);
  // up i=1 -> N/4
  k_upadd<<<((N / 4) * FH + 255) / 256, 256, 0, stream>>>(cur, D2, alt, N / 4);
  { float* t = cur; cur = alt; alt = t; }
  lap_block(cur, N / 4, 1, 8, g_up_b + 0 * FH);
  // up i=2 -> N/2
  k_upadd<<<((N / 2) * FH + 255) / 256, 256, 0, stream>>>(cur, D1, alt, N / 2);
  { float* t = cur; cur = alt; alt = t; }
  lap_block(cur, N / 2, 2, 10, g_up_b + 2 * FH);
  // up i=3 -> N
  k_upadd<<<(N * FH + 255) / 256, 256, 0, stream>>>(cur, D0, alt, N);
  { float* t = cur; cur = alt; alt = t; }
  lap_block(cur, N, 3, 12, g_up_b + 4 * FH);
  k_head_g<<<(N * 2 + 255) / 256, 256, 0, stream>>>(cur, g_conv2_W, g_conv2_b, inputs, SWG, N);

  // ================= local branch =================
  float* Y = alt;  // free buffer after global branch
  k_conv_in<<<(N * FH + 255) / 256, 256, 0, stream>>>(inputs, l_conv1_W, l_conv1_b, Y, N);
  k_mask_sum<<<1, 256, 0, stream>>>(mask1, DEN, N);
  for (int i = 0; i < 15; ++i) {
    int w = 14 + 2 * i;
    const float* bb = l_rn_b + (size_t)(2 * i) * FH;
    hipMemcpyAsync(INP, Y, (size_t)N * FH * 4, hipMemcpyDeviceToDevice, stream);
    if (i % 2 == 0) {
      lap_layer(Y, N, L_rows, L_cols, L_vals, L_ne, w, bb, nullptr);
      lap_layer(Y, N, L_rows, L_cols, L_vals, L_ne, w + 1, bb + FH, INP);
    } else {
      avg_layer(Y, N, w, bb, nullptr);
      avg_layer(Y, N, w + 1, bb + FH, INP);
    }
  }
  k_head_l<<<(N + 255) / 256, 256, 0, stream>>>(Y, l_conv2_W, l_conv2_b, inputs, SL, N);
  k_final<<<(N + 255) / 256, 256, 0, stream>>>(SWG, SL, (float*)d_out, N);
}
